// MultiHeadAttention_65446711656594
// MI455X (gfx1250) — matrix operand order repaired and checked by reading, not compiled, not run on hardware
//
#include <hip/hip_runtime.h>

// ---------------------------------------------------------------------------
// MHA for gfx1250 (MI455X): f16 WMMA + flash-attention online softmax.
// B=2, S=2048, D=1024, H=16, hd=64. wave32 only.
// 4 launches: convert | fused QKV GEMM | flash attention | output projection.
// ---------------------------------------------------------------------------

#define BATCH 2
#define SEQ   2048
#define DMODEL 1024
#define HEADS 16
#define HDIM  64

typedef _Float16 v8h  __attribute__((ext_vector_type(8)));
typedef _Float16 v16h __attribute__((ext_vector_type(16)));
typedef float    v8f  __attribute__((ext_vector_type(8)));
typedef int      v4i  __attribute__((ext_vector_type(4)));

#if defined(__HIP_DEVICE_COMPILE__) && __has_builtin(__builtin_amdgcn_global_load_async_to_lds_b128)
#define HAS_ASYNC_LDS 1
#else
#define HAS_ASYNC_LDS 0
#endif

// ---- WMMA fragment loaders (contiguous-K order shared by A and B, wave32) ----------

// A-matrix 16x32 f16: lane holds row (lane&15); 16 contiguous K values
// starting at 16*(lane>>4) (same K order as load_bmat).
__device__ __forceinline__ v16h load_amat(const _Float16* base, int ld, int lane) {
  const int row = lane & 15;
  const int kh  = (lane >> 4) << 4;
  const _Float16* p = base + row * ld + kh;
  v8h a = *(const v8h*)(p);
  v8h b = *(const v8h*)(p + 8);
  return __builtin_shufflevector(a, b, 0,1,2,3,4,5,6,7,8,9,10,11,12,13,14,15);
}

// B-matrix 32x16 f16: lane holds column (lane&15); 16 contiguous K values
// starting at 16*(lane>>4).  Source is "B^T" row-major: [N][K], ld = K-stride.
__device__ __forceinline__ v16h load_bmat(const _Float16* base, int ld, int lane) {
  const int col = lane & 15;
  const int kb  = (lane >> 4) << 4;
  const _Float16* p = base + col * ld + kb;
  v8h a = *(const v8h*)(p);
  v8h b = *(const v8h*)(p + 8);
  return __builtin_shufflevector(a, b, 0,1,2,3,4,5,6,7,8,9,10,11,12,13,14,15);
}

__device__ __forceinline__ v8f wmma_f16(v16h a, v16h b, v8f c) {
  return __builtin_amdgcn_wmma_f32_16x16x32_f16(false, a, false, b,
                                                (short)0, c, false, false);
}

#if HAS_ASYNC_LDS
// AS-qualified pointer types per clang's diagnosed builtin signature:
// param0 = <4 x i32> addrspace(1)*, param1 = <4 x i32> addrspace(3)*.
typedef __attribute__((address_space(1))) v4i* gv4i_p;
typedef __attribute__((address_space(3))) v4i* lv4i_p;

__device__ __forceinline__ void async_cp16(const _Float16* g, _Float16* l) {
  __builtin_amdgcn_global_load_async_to_lds_b128((gv4i_p)g, (lv4i_p)l, 0, 0);
}
// Stage one V^T tile: 64 hd-rows x 32 keys (ld SEQ in global) -> [64][32] LDS.
// 4 KiB = 256 x 16B chunks = 8 async b128 per lane.
__device__ __forceinline__ void stage_v_tile(const _Float16* Vb, int k0,
                                             _Float16* buf, int lane) {
#pragma unroll
  for (int i = 0; i < 8; ++i) {
    const int c   = i * 32 + lane;       // 16B chunk id (4 chunks per 64B row)
    const int row = c >> 2;
    const int off = (c & 3) << 3;        // halves within row
    async_cp16(Vb + (size_t)row * SEQ + k0 + off, buf + row * 32 + off);
  }
}
__device__ __forceinline__ void wait_async_8() {
#if __has_builtin(__builtin_amdgcn_s_wait_asynccnt)
  __builtin_amdgcn_s_wait_asynccnt(8);
#else
  asm volatile("s_wait_asynccnt 0x8" ::: "memory");
#endif
  asm volatile("" ::: "memory");
}
__device__ __forceinline__ void wait_async_0() {
#if __has_builtin(__builtin_amdgcn_s_wait_asynccnt)
  __builtin_amdgcn_s_wait_asynccnt(0);
#else
  asm volatile("s_wait_asynccnt 0x0" ::: "memory");
#endif
  asm volatile("" ::: "memory");
}
#endif

// ---- fp32 -> fp16 conversion: one launch for x + all four weights ----------
// Destinations are contiguous in workspace: [xh | Wqh | Wkh | Wvh | Woh].
__global__ void cvt_all_kernel(const float* __restrict__ x,
                               const float* __restrict__ Wq,
                               const float* __restrict__ Wk,
                               const float* __restrict__ Wv,
                               const float* __restrict__ Wo,
                               _Float16* __restrict__ dst) {
  const int NX = BATCH * SEQ * DMODEL;      // 4M
  const int NW = DMODEL * DMODEL;           // 1M
  const int i = blockIdx.x * blockDim.x + threadIdx.x;
  if (i >= NX + 4 * NW) return;
  const float* src;
  int off;
  if (i < NX) { src = x; off = i; }
  else {
    const int j = i - NX;
    const int w = j >> 20;                  // / NW
    off = j & (NW - 1);
    src = (w == 0) ? Wq : (w == 1) ? Wk : (w == 2) ? Wv : Wo;
  }
  dst[i] = (_Float16)src[off];
}

// ---- Fused QKV projection: wave = 16x64 tile of Q, K and V -----------------
// A-fragment of x loaded once, feeds 12 WMMAs per 32-wide K chunk.
// Q,K out [B,H,S,hd] f16; V out transposed [B,H,hd,S] f16.
__global__ void qkv_wmma_kernel(const _Float16* __restrict__ A,
                                const _Float16* __restrict__ Wqh,
                                const _Float16* __restrict__ Wkh,
                                const _Float16* __restrict__ Wvh,
                                const float* __restrict__ bq,
                                const float* __restrict__ bk,
                                const float* __restrict__ bv,
                                _Float16* __restrict__ Qh,
                                _Float16* __restrict__ Kh,
                                _Float16* __restrict__ Vth) {
  const int lane = threadIdx.x & 31;
  const int wv   = threadIdx.x >> 5;
  const int gw   = blockIdx.x * 8 + wv;
  if (gw >= (BATCH * SEQ / 16) * (DMODEL / 64)) return;
  const int mt = gw >> 4;                       // token tile (256)
  const int nt = gw & 15;                       // 64-wide N tile (16)

  v8f acq[4] = {}, ack[4] = {}, acv[4] = {};
  const _Float16* Arow = A + (size_t)mt * 16 * DMODEL;

  for (int kc = 0; kc < DMODEL; kc += 32) {
    __builtin_prefetch(Arow + kc + 64, 0, 1);
    v16h a = load_amat(Arow + kc, DMODEL, lane);
#pragma unroll
    for (int j = 0; j < 4; ++j) {
      const size_t boff = (size_t)(nt * 64 + j * 16) * DMODEL + kc;
      acq[j] = wmma_f16(a, load_bmat(Wqh + boff, DMODEL, lane), acq[j]);
      ack[j] = wmma_f16(a, load_bmat(Wkh + boff, DMODEL, lane), ack[j]);
      acv[j] = wmma_f16(a, load_bmat(Wvh + boff, DMODEL, lane), acv[j]);
    }
  }

  const int colq = lane & 15;
  const int rofs = (lane >> 4) << 3;            // C layout: row = r + 8*(lane>>4)
#pragma unroll
  for (int j = 0; j < 4; ++j) {
    const int col = nt * 64 + j * 16 + colq;
    const float bqv = bq[col], bkv = bk[col], bvv = bv[col];
    const int h = col >> 6, hd = col & (HDIM - 1);
#pragma unroll
    for (int r = 0; r < 8; ++r) {
      const int row = mt * 16 + r + rofs;
      const int b = row >> 11, s = row & (SEQ - 1);
      const size_t bh = (size_t)(b * HEADS + h);
      Qh[(bh * SEQ + s) * HDIM + hd]  = (_Float16)(acq[j][r] + bqv);
      Kh[(bh * SEQ + s) * HDIM + hd]  = (_Float16)(ack[j][r] + bkv);
      Vth[(bh * HDIM + hd) * SEQ + s] = (_Float16)(acv[j][r] + bvv);
    }
  }
}

// ---- Flash attention: one wave per (b, h, 16-query tile) -------------------
// Q,K in [B,H,S,hd] f16; V in [B,H,hd,S] f16; ctx out [B,S,D] f16.
__global__ void attn_wmma_kernel(const _Float16* __restrict__ Q,
                                 const _Float16* __restrict__ Kt,
                                 const _Float16* __restrict__ Vt,
                                 const int* __restrict__ mask,
                                 _Float16* __restrict__ ctx) {
  __shared__ __align__(16) _Float16 lds_p[8][16 * 32];      // per-wave P tile
#if HAS_ASYNC_LDS
  __shared__ __align__(16) _Float16 lds_v[8][2][64 * 32];   // per-wave V dbl buf
#endif

  const int lane = threadIdx.x & 31;
  const int wv   = threadIdx.x >> 5;
  const int task = blockIdx.x * 8 + wv;
  if (task >= BATCH * HEADS * (SEQ / 16)) return;
  const int b  = task >> 11;
  const int h  = (task >> 7) & (HEADS - 1);
  const int qt = task & (SEQ / 16 - 1);

  const _Float16* Qb = Q + (((size_t)(b * HEADS + h) * SEQ) + qt * 16) * HDIM;
  const _Float16* Kb = Kt + (size_t)(b * HEADS + h) * SEQ * HDIM;
  const _Float16* Vb = Vt + (size_t)(b * HEADS + h) * HDIM * SEQ;
  const int* mb = mask + b * SEQ;

  v16h aq[2];
  aq[0] = load_amat(Qb, HDIM, lane);
  aq[1] = load_amat(Qb + 32, HDIM, lane);

  v8f o[4] = {};
  float mrow[8], lrow[8];
#pragma unroll
  for (int r = 0; r < 8; ++r) { mrow[r] = -1e30f; lrow[r] = 0.0f; }

  const int colq = lane & 15;

#if HAS_ASYNC_LDS
  stage_v_tile(Vb, 0, &lds_v[wv][0][0], lane);    // prologue: tile 0 in flight
#endif

  for (int kt = 0; kt < SEQ / 32; ++kt) {
    const int k0 = kt * 32;
#if HAS_ASYNC_LDS
    const int cur = kt & 1;
    if (kt + 1 < SEQ / 32)
      stage_v_tile(Vb, k0 + 32, &lds_v[wv][cur ^ 1][0], lane);
#endif

    // scores: S[16q x 32k] via 4 WMMAs (2 key tiles x 2 hd chunks)
    v8f s0 = {}, s1 = {};
#pragma unroll
    for (int c = 0; c < 2; ++c) {
      v16h bk0 = load_bmat(Kb + (size_t)k0 * HDIM + c * 32, HDIM, lane);
      s0 = wmma_f16(aq[c], bk0, s0);
      v16h bk1 = load_bmat(Kb + (size_t)(k0 + 16) * HDIM + c * 32, HDIM, lane);
      s1 = wmma_f16(aq[c], bk1, s1);
    }

    const float madd0 = (mb[k0 + colq] == 0) ? -1e9f : 0.0f;
    const float madd1 = (mb[k0 + 16 + colq] == 0) ? -1e9f : 0.0f;
#pragma unroll
    for (int r = 0; r < 8; ++r) {
      s0[r] = s0[r] * 0.125f + madd0;             // 1/sqrt(64)
      s1[r] = s1[r] * 0.125f + madd1;
    }

    // online softmax; C-layout rows live in 16-lane halves -> half-wave shfl.
#pragma unroll
    for (int r = 0; r < 8; ++r) {
      float mx = fmaxf(s0[r], s1[r]);
#pragma unroll
      for (int off = 8; off; off >>= 1) mx = fmaxf(mx, __shfl_xor(mx, off, 32));
      const float mn = fmaxf(mrow[r], mx);
      const float sc = __expf(mrow[r] - mn);
      mrow[r] = mn;
      const float p0 = __expf(s0[r] - mn);
      const float p1 = __expf(s1[r] - mn);
      float rs = p0 + p1;
#pragma unroll
      for (int off = 8; off; off >>= 1) rs += __shfl_xor(rs, off, 32);
      lrow[r] = lrow[r] * sc + rs;
      o[0][r] *= sc; o[1][r] *= sc; o[2][r] *= sc; o[3][r] *= sc;
      const int row = r + ((lane >> 4) << 3);
      lds_p[wv][row * 32 + colq]      = (_Float16)p0;
      lds_p[wv][row * 32 + 16 + colq] = (_Float16)p1;
    }
    // Intra-wave LDS ops retire in order (ISA 7.3) -> only a compiler fence.
    asm volatile("" ::: "memory");

    // P[16x32] as A-fragment, then ctx += P x V.
    v16h ap = load_amat(&lds_p[wv][0], 32, lane);
#if HAS_ASYNC_LDS
    // async loads complete in order: <=8 outstanding means current tile landed.
    if (kt + 1 < SEQ / 32) wait_async_8(); else wait_async_0();
    const _Float16* vsrc = &lds_v[wv][cur][0];
#pragma unroll
    for (int jj = 0; jj < 4; ++jj) {
      v16h bv = load_bmat(vsrc + (size_t)(jj * 16) * 32, 32, lane);
      o[jj] = wmma_f16(ap, bv, o[jj]);
    }
#else
#pragma unroll
    for (int jj = 0; jj < 4; ++jj) {
      v16h bv = load_bmat(Vb + (size_t)(jj * 16) * SEQ + k0, SEQ, lane);
      o[jj] = wmma_f16(ap, bv, o[jj]);
    }
#endif
  }

  // Normalize and write ctx in [B,S,D] f16 (heads concatenated).
  _Float16* cb = ctx + ((size_t)(b * SEQ) + qt * 16) * DMODEL + h * HDIM;
#pragma unroll
  for (int r = 0; r < 8; ++r) {
    const float inv = 1.0f / lrow[r];
    const int row = r + ((lane >> 4) << 3);
#pragma unroll
    for (int jj = 0; jj < 4; ++jj)
      cb[(size_t)row * DMODEL + jj * 16 + colq] = (_Float16)(o[jj][r] * inv);
  }
}

// ---- Output projection: out[m][n] = sum_k C[m][k]*Wo[n][k] + bo[n] (f32) ---
__global__ void oproj_wmma_kernel(const _Float16* __restrict__ A,
                                  const _Float16* __restrict__ Bt,
                                  const float* __restrict__ bias,
                                  float* __restrict__ out) {
  const int lane = threadIdx.x & 31;
  const int wv   = threadIdx.x >> 5;
  const int gw   = blockIdx.x * 8 + wv;
  if (gw >= (BATCH * SEQ / 16) * (DMODEL / 64)) return;
  const int mt = gw >> 4;
  const int nt = gw & 15;

  v8f acc[4] = {};
  const _Float16* Arow = A + (size_t)mt * 16 * DMODEL;
  const _Float16* Brow = Bt + (size_t)nt * 64 * DMODEL;

  for (int kc = 0; kc < DMODEL; kc += 32) {
    __builtin_prefetch(Arow + kc + 64, 0, 1);
    v16h a = load_amat(Arow + kc, DMODEL, lane);
#pragma unroll
    for (int j = 0; j < 4; ++j) {
      v16h b = load_bmat(Brow + (size_t)(j * 16) * DMODEL + kc, DMODEL, lane);
      acc[j] = wmma_f16(a, b, acc[j]);
    }
  }

  const int colq = lane & 15;
  const int rofs = (lane >> 4) << 3;
#pragma unroll
  for (int j = 0; j < 4; ++j) {
    const int col = nt * 64 + j * 16 + colq;
    const float bv = bias[col];
#pragma unroll
    for (int r = 0; r < 8; ++r) {
      const int row = mt * 16 + r + rofs;
      out[(size_t)row * DMODEL + col] = acc[j][r] + bv;
    }
  }
}

// ---------------------------------------------------------------------------

extern "C" void kernel_launch(void* const* d_in, const int* in_sizes, int n_in,
                              void* d_out, int out_size, void* d_ws, size_t ws_size,
                              hipStream_t stream) {
  const float* x    = (const float*)d_in[0];
  const int*   mask = (const int*)d_in[1];
  const float* Wq = (const float*)d_in[2];
  const float* bq = (const float*)d_in[3];
  const float* Wk = (const float*)d_in[4];
  const float* bk = (const float*)d_in[5];
  const float* Wv = (const float*)d_in[6];
  const float* bv = (const float*)d_in[7];
  const float* Wo = (const float*)d_in[8];
  const float* bo = (const float*)d_in[9];

  const int M = BATCH * SEQ;            // 4096
  const int N = DMODEL;                 // 1024
  const int K = DMODEL;                 // 1024

  char* ws = (char*)d_ws;
  size_t off = 0;
  _Float16* xh  = (_Float16*)(ws + off); off += (size_t)M * K * 2;   // contiguous:
  _Float16* Wqh = (_Float16*)(ws + off); off += (size_t)N * K * 2;   // [xh|Wqh|Wkh|Wvh|Woh]
  _Float16* Wkh = (_Float16*)(ws + off); off += (size_t)N * K * 2;
  _Float16* Wvh = (_Float16*)(ws + off); off += (size_t)N * K * 2;
  _Float16* Woh = (_Float16*)(ws + off); off += (size_t)N * K * 2;
  _Float16* Qh  = (_Float16*)(ws + off); off += (size_t)M * N * 2;
  _Float16* Kh  = (_Float16*)(ws + off); off += (size_t)M * N * 2;
  _Float16* Vth = (_Float16*)(ws + off); off += (size_t)M * N * 2;
  _Float16* Ch  = (_Float16*)(ws + off); off += (size_t)M * N * 2;   // 48 MiB total

  // 1) fp32 -> fp16, single launch (x + 4 weight matrices)
  {
    const int ntot = M * K + 4 * N * K;                 // 8,388,608
    cvt_all_kernel<<<(ntot + 255) / 256, 256, 0, stream>>>(x, Wq, Wk, Wv, Wo, xh);
  }

  const int waves  = (M / 16) * (N / 64);               // 4096
  const int blocks = waves / 8;                         // 512

  // 2) fused Q/K/V projection (A-fragment of x feeds 12 WMMAs per chunk)
  qkv_wmma_kernel<<<blocks, 256, 0, stream>>>(xh, Wqh, Wkh, Wvh, bq, bk, bv,
                                              Qh, Kh, Vth);

  // 3) flash attention (wave = one 16-query tile of one head)
  attn_wmma_kernel<<<blocks, 256, 0, stream>>>(Qh, Kh, Vth, mask, Ch);

  // 4) output projection -> d_out (fp32)
  oproj_wmma_kernel<<<blocks, 256, 0, stream>>>(Ch, Woh, bo, (float*)d_out);
}
